// simpleHeteroGATConv_1623497638552
// MI455X (gfx1250) — matrix operand order repaired and checked by reading, not compiled, not run on hardware
//
#include <hip/hip_runtime.h>
#include <math.h>

#define IN_DIM 128
#define H 4
#define D 16
#define HD 64      // H*D
#define EF 32
#define T 8
#define NEG_SLOPE 0.2f

typedef __attribute__((ext_vector_type(16))) _Float16 v16h;
typedef __attribute__((ext_vector_type(8)))  _Float16 v8h;
typedef __attribute__((ext_vector_type(8)))  float    v8f;

union V16U { v16h v; v8h h2[2]; _Float16 e[16]; };

// ---------------------------------------------------------------------------
// feat = x @ fc_W  via v_wmma_f32_16x16x32_f16.
// Block = 256 threads = 8 waves; each wave -> 16 rows x 64 cols output tile.
// fc_W staged in LDS as f16, transposed [col][k], padded to avoid conflicts.
// ---------------------------------------------------------------------------
#define LDSTRIDE 136   // 128 + 8 halves pad

__global__ __launch_bounds__(256) void gemm_feat_kernel(
    const float* __restrict__ x, const float* __restrict__ W,
    float* __restrict__ feat, int N)
{
  __shared__ __align__(16) _Float16 whT[64 * LDSTRIDE];

  // cooperative load + transpose + f32->f16 convert of W [128,64]
  for (int i = threadIdx.x; i < IN_DIM * HD; i += 256) {
    int k = i >> 6;   // 0..127
    int c = i & 63;   // 0..63
    whT[c * LDSTRIDE + k] = (_Float16)W[i];
  }
  __syncthreads();

  const int wave = threadIdx.x >> 5;
  const int lane = threadIdx.x & 31;
  const int hi   = lane >> 4;          // which 16-lane half
  const int m0   = blockIdx.x * 128 + wave * 16;

  int mrow = m0 + (lane & 15);
  int mcl  = mrow < N ? mrow : (N - 1);          // clamp: keep EXEC all-1s
  const float* __restrict__ arow = x + (size_t)mcl * IN_DIM;

  v8f acc[4] = {v8f{}, v8f{}, v8f{}, v8f{}};

  for (int kc = 0; kc < IN_DIM; kc += 32) {
    // A fragment: 16-bit A 16x32 layout (lane = M; VGPR j holds K pairs)
    V16U a;
#pragma unroll
    for (int j = 0; j < 8; ++j) {
      int kb = kc + ((j < 4) ? 2 * j : 16 + 2 * (j - 4)) + (hi ? 8 : 0);
      float2 f2 = *(const float2*)(arow + kb);
      a.e[2 * j]     = (_Float16)f2.x;
      a.e[2 * j + 1] = (_Float16)f2.y;
    }
    const int kbase = kc + (hi ? 8 : 0);
#pragma unroll
    for (int nt = 0; nt < 4; ++nt) {
      int col = nt * 16 + (lane & 15);
      const v8h* bp = (const v8h*)(whT + col * LDSTRIDE + kbase);
      V16U b;
      b.h2[0] = bp[0];
      b.h2[1] = bp[2];
      acc[nt] = __builtin_amdgcn_wmma_f32_16x16x32_f16(
          false, a.v, false, b.v, (short)0, acc[nt], false, false);
    }
  }

  // C/D layout: VGPR r -> row m0 + r (+8 for upper lane half), col = lane&15
#pragma unroll
  for (int nt = 0; nt < 4; ++nt) {
#pragma unroll
    for (int r = 0; r < 8; ++r) {
      int row = m0 + r + (hi ? 8 : 0);
      if (row < N)
        feat[(size_t)row * HD + nt * 16 + (lane & 15)] = acc[nt][r];
    }
  }
}

// ---------------------------------------------------------------------------
// ee[t,h] = sum_e (edge_emb @ fc_e_W)[t,h,e] * attn_e[h,e]   (32 outputs)
// ---------------------------------------------------------------------------
__global__ void ee_kernel(const float* __restrict__ edge_emb,
                          const float* __restrict__ fc_e_W,
                          const float* __restrict__ attn_e,
                          float* __restrict__ ee)
{
  int i = threadIdx.x;
  if (i >= T * H) return;
  int t = i >> 2, h = i & 3;
  float acc = 0.f;
  for (int e = 0; e < EF; ++e) {
    float f = 0.f;
    for (int k = 0; k < EF; ++k)
      f += edge_emb[t * EF + k] * fc_e_W[k * (H * EF) + h * EF + e];
    acc += f * attn_e[h * EF + e];
  }
  ee[t * H + h] = acc;
}

// ---------------------------------------------------------------------------
// el/er: per-node per-head dot of feat row with attn_l/attn_r
// ---------------------------------------------------------------------------
__global__ void elr_kernel(const float* __restrict__ feat,
                           const float* __restrict__ attn_l,
                           const float* __restrict__ attn_r,
                           float* __restrict__ el, float* __restrict__ er, int N)
{
  int n = blockIdx.x * blockDim.x + threadIdx.x;
  if (n >= N) return;
  const float* f = feat + (size_t)n * HD;
#pragma unroll
  for (int h = 0; h < H; ++h) {
    float sl = 0.f, sr = 0.f;
#pragma unroll
    for (int d = 0; d < D; ++d) {
      float v = f[h * D + d];
      sl += v * attn_l[h * D + d];
      sr += v * attn_r[h * D + d];
    }
    el[n * H + h] = sl;
    er[n * H + h] = sr;
  }
}

// ---------------------------------------------------------------------------
// init: zero out & denom, -inf smax
// ---------------------------------------------------------------------------
__global__ void init_kernel(float* __restrict__ out, float* __restrict__ smax,
                            float* __restrict__ denom, int n64, int n4)
{
  int i = blockIdx.x * blockDim.x + threadIdx.x;
  if (i < n64) out[i] = 0.f;
  if (i < n4) { smax[i] = -INFINITY; denom[i] = 0.f; }
}

// ---------------------------------------------------------------------------
// per-(edge,head) leaky-relu logits + segment max via int-atomic trick
// ---------------------------------------------------------------------------
__global__ void logits_kernel(const int* __restrict__ src,
                              const int* __restrict__ dst,
                              const int* __restrict__ etype,
                              const float* __restrict__ el,
                              const float* __restrict__ er,
                              const float* __restrict__ ee,
                              float* __restrict__ s, float* __restrict__ smax,
                              int E)
{
  int i = blockIdx.x * blockDim.x + threadIdx.x;
  if (i >= E * H) return;
  int e = i >> 2, h = i & 3;
  float v = el[src[e] * H + h] + er[dst[e] * H + h] + ee[etype[e] * H + h];
  v = v > 0.f ? v : NEG_SLOPE * v;
  s[i] = v;
  float* addr = &smax[dst[e] * H + h];
  if (v >= 0.f) atomicMax((int*)addr, __float_as_int(v));
  else          atomicMin((unsigned int*)addr, __float_as_uint(v));
}

// ---------------------------------------------------------------------------
// p = exp(s - smax[dst]);  denom[dst] += p  (p overwrites s in place)
// ---------------------------------------------------------------------------
__global__ void expsum_kernel(const int* __restrict__ dst,
                              float* __restrict__ s,
                              const float* __restrict__ smax,
                              float* __restrict__ denom, int E)
{
  int i = blockIdx.x * blockDim.x + threadIdx.x;
  if (i >= E * H) return;
  int e = i >> 2, h = i & 3;
  float p = __expf(s[i] - smax[dst[e] * H + h]);
  s[i] = p;
  atomicAdd(&denom[dst[e] * H + h], p);
}

// ---------------------------------------------------------------------------
// out[dst,h,:] += (p/denom) * feat[src,h,:]   — 16 contiguous f32 atomics,
// working set fits in 192MB L2 so scattered atomics stay on-chip.
// ---------------------------------------------------------------------------
__global__ void agg_kernel(const int* __restrict__ src,
                           const int* __restrict__ dst,
                           const float* __restrict__ p,
                           const float* __restrict__ denom,
                           const float* __restrict__ feat,
                           float* __restrict__ out, int E)
{
  int i = blockIdx.x * blockDim.x + threadIdx.x;
  if (i >= E * H) return;
  int e = i >> 2, h = i & 3;
  int de = dst[e], sr = src[e];
  float a = p[i] / denom[de * H + h];
  const float* f = feat + (size_t)sr * HD + h * D;
  float* o = out + (size_t)de * HD + h * D;
#pragma unroll
  for (int d = 0; d < D; ++d) atomicAdd(&o[d], f[d] * a);
}

// ---------------------------------------------------------------------------
extern "C" void kernel_launch(void* const* d_in, const int* in_sizes, int n_in,
                              void* d_out, int out_size, void* d_ws, size_t ws_size,
                              hipStream_t stream)
{
  const float* x        = (const float*)d_in[0];
  const int*   src      = (const int*)d_in[1];
  const int*   dst      = (const int*)d_in[2];
  const int*   etype    = (const int*)d_in[3];
  const float* fc_W     = (const float*)d_in[4];
  const float* fc_e_W   = (const float*)d_in[5];
  const float* edge_emb = (const float*)d_in[6];
  const float* attn_l   = (const float*)d_in[7];
  const float* attn_r   = (const float*)d_in[8];
  const float* attn_e   = (const float*)d_in[9];

  const int N = in_sizes[0] / IN_DIM;
  const int E = in_sizes[1];

  char* ws = (char*)d_ws;
  size_t off = 0;
  auto alloc = [&](size_t bytes) -> void* {
    void* p = ws + off;
    off += (bytes + 255) & ~(size_t)255;
    return p;
  };
  float* feat  = (float*)alloc((size_t)N * HD * sizeof(float));
  float* el    = (float*)alloc((size_t)N * H * sizeof(float));
  float* er    = (float*)alloc((size_t)N * H * sizeof(float));
  float* ee    = (float*)alloc((size_t)T * H * sizeof(float));
  float* smax  = (float*)alloc((size_t)N * H * sizeof(float));
  float* denom = (float*)alloc((size_t)N * H * sizeof(float));
  float* s     = (float*)alloc((size_t)E * H * sizeof(float));

  float* out = (float*)d_out;

  // init out / smax / denom
  {
    int n64 = N * HD, n4 = N * H;
    int grid = (n64 + 255) / 256;
    init_kernel<<<grid, 256, 0, stream>>>(out, smax, denom, n64, n4);
  }
  // WMMA node projection
  {
    int grid = (N + 127) / 128;
    gemm_feat_kernel<<<grid, 256, 0, stream>>>(x, fc_W, feat, N);
  }
  // tiny per-etype scalar
  ee_kernel<<<1, 32, 0, stream>>>(edge_emb, fc_e_W, attn_e, ee);
  // per-node attention dots
  elr_kernel<<<(N + 255) / 256, 256, 0, stream>>>(feat, attn_l, attn_r, el, er, N);
  // edge phase
  {
    int grid = (E * H + 255) / 256;
    logits_kernel<<<grid, 256, 0, stream>>>(src, dst, etype, el, er, ee, s, smax, E);
    expsum_kernel<<<grid, 256, 0, stream>>>(dst, s, smax, denom, E);
    agg_kernel<<<grid, 256, 0, stream>>>(src, dst, s, denom, feat, out, E);
  }
}
